// Net_40664750358958
// MI455X (gfx1250) — matrix operand order repaired and checked by reading, not compiled, not run on hardware
//
#include <hip/hip_runtime.h>
#include <hip/hip_bf16.h>

// ---------------------------------------------------------------------------
// Types / helpers
// ---------------------------------------------------------------------------
typedef __attribute__((ext_vector_type(16))) __bf16       v16bf;
typedef __attribute__((ext_vector_type(8)))  float        v8f;
typedef __attribute__((ext_vector_type(8)))  unsigned int v8u;
typedef __attribute__((ext_vector_type(4)))  unsigned int v4u;
typedef __attribute__((ext_vector_type(8)))  int          v8i;
typedef __attribute__((ext_vector_type(4)))  int          v4i;

#define DEV __device__ __forceinline__

DEV unsigned short f2bf(float f) {           // round-to-nearest-even f32->bf16
  unsigned u = __builtin_bit_cast(unsigned, f);
  u += 0x7FFFu + ((u >> 16) & 1u);
  return (unsigned short)(u >> 16);
}
DEV float bf2f(unsigned short h) {
  return __builtin_bit_cast(float, ((unsigned)h) << 16);
}

// ---------------------------------------------------------------------------
// Tensor Data Mover: load a 256-row x 32-col bf16 tile of a [256][K] matrix
// into LDS with 4-dword padding per 16-dword row => LDS row stride 40 elems.
// D# layout per CDNA5 ISA sec 8.3/8.4.
// ---------------------------------------------------------------------------
#if defined(__has_builtin)
#if __has_builtin(__builtin_amdgcn_tensor_load_to_lds) && __has_builtin(__builtin_amdgcn_s_wait_tensorcnt)
#define HAVE_TDM 1
#endif
#endif
#ifndef HAVE_TDM
#define HAVE_TDM 0
#endif

#if HAVE_TDM
DEV void tdm_load_tile_A(unsigned lds_byte_off, const unsigned short* gsrc,
                         unsigned Kelems) {
  unsigned long long ga = (unsigned long long)gsrc;
  v4u g0;
  g0[0] = 1u;                                   // count=1, user descriptor
  g0[1] = lds_byte_off;                         // lds_addr [63:32]
  g0[2] = (unsigned)ga;                         // global_addr lo
  g0[3] = (unsigned)(ga >> 32) | (2u << 30);    // global_addr hi | type=2
  const unsigned td0 = Kelems;                  // tensor_dim0 (row length, elems)
  const unsigned td1 = 256u;                    // tensor_dim1 (rows)
  const unsigned tile0 = 32u, tile1 = 256u;     // tile: 32 cols x 256 rows
  // data_size=1 (2B) | pad_enable | pad_interval=3 (16 dw) | pad_amount=3 (4 dw)
  const unsigned d0 = (1u << 16) | (1u << 20) | (3u << 22) | (3u << 25);
  v8i g1;
  g1[0] = (int)d0;
  g1[1] = (int)((td0 & 0xFFFFu) << 16);                       // tensor_dim0 lo16
  g1[2] = (int)((td0 >> 16) | ((td1 & 0xFFFFu) << 16));       // dim0 hi | dim1 lo
  g1[3] = (int)((td1 >> 16) | (tile0 << 16));                 // dim1 hi | tile_dim0
  g1[4] = (int)tile1;                                         // tile_dim1, tile_dim2=0
  g1[5] = (int)Kelems;                                        // tensor_dim0_stride lo32
  g1[6] = 0;                                                  // stride0 hi | stride1 lo
  g1[7] = 0;
  v4i z4 = {0, 0, 0, 0};
#if __clang_major__ >= 23
  v8i z8 = {0, 0, 0, 0, 0, 0, 0, 0};
  __builtin_amdgcn_tensor_load_to_lds(g0, g1, z4, z4, z8, 0);
#else
  __builtin_amdgcn_tensor_load_to_lds(g0, g1, z4, z4, 0);
#endif
}
#endif

// ---------------------------------------------------------------------------
// Elementwise converts
// ---------------------------------------------------------------------------
__global__ void cvt_bf16_kernel(const float* __restrict__ in,
                                unsigned short* __restrict__ out, int n) {
  int i = blockIdx.x * 256 + threadIdx.x;
  if (i < n) out[i] = f2bf(in[i]);
}

// A1: conv1 weights [256][81] -> bf16 [256][96] zero-padded in K (k=ky*9+kx)
__global__ void build_A1_kernel(const float* __restrict__ w,
                                unsigned short* __restrict__ A1) {
  int i = blockIdx.x * 256 + threadIdx.x;
  if (i < 256 * 96) {
    int m = i / 96, k = i % 96;
    A1[i] = (k < 81) ? f2bf(w[m * 81 + k]) : (unsigned short)0;
  }
}

// A2: conv2 weights [co][ci][ky][kx] -> bf16 [co][(ky*9+kx)*256 + ci]
// (K-permutation so a 32-wide K tile has uniform (ky,kx) and contiguous ci)
__global__ void build_A2_kernel(const float* __restrict__ w,
                                unsigned short* __restrict__ A2) {
  int o = blockIdx.x * 256 + threadIdx.x;
  if (o < 256 * 20736) {
    int m = o / 20736, q = o % 20736;
    int ci = q & 255, kykx = q >> 8;
    A2[o] = f2bf(w[(size_t)(m * 256 + ci) * 81 + kykx]);
  }
}

// ---------------------------------------------------------------------------
// Tiled bf16 WMMA GEMM with implicit im2col B gather.
// MODE 0: conv1  (M=256, K=96(pad 81), N=204800), epilogue bias+ReLU -> bf16 h
// MODE 1: conv2  (M=256, K=20736 permuted, N=18432), epilogue bias   -> f32 p
// Block = 256 threads = 8 waves; block tile 256(M) x 64(N); wave tile 64x32.
// ---------------------------------------------------------------------------
template <int MODE>
__global__ __launch_bounds__(256, 2) void gemm_wmma_kernel(
    const unsigned short* __restrict__ A,     // [256][K] bf16 row-major
    const unsigned short* __restrict__ Bsrc,  // gather source (x_bf16 or h_bf16)
    const float* __restrict__ bias, void* __restrict__ Out) {
  constexpr int K = (MODE == 0) ? 96 : 20736;
  constexpr int NSTEPS = K / 32;

  __shared__ unsigned short lA[2][256 * 40];  // row stride 40 (TDM-padded)
  __shared__ unsigned short lB[2][64 * 34];   // [n][k], stride 34

  const int tid = threadIdx.x;
  const int lane = tid & 31;
  const int wave = tid >> 5;
  const int n0 = blockIdx.x * 64;

  // Per-thread gather constants for this thread's B column.
  const int nloc = tid & 63;  // column within tile
  // k sub-group: tid>>6 is wave-uniform (wave w -> w>>1); force into an SGPR so
  // the whole k -> offset computation scalarizes (SALU + saddr-form loads).
  const int khs = __builtin_amdgcn_readfirstlane(tid >> 6);
  int cN;
  {
    const int n = n0 + nloc;
    if (MODE == 0) {
      int bb = n / 400, rr = n % 400;
      int y = rr / 20, x = rr % 20;
      cN = bb * 784 + y * 28 + x;                   // + ky*28 + kx
    } else {
      int bb = n / 36, r2 = n % 36;
      int y = r2 / 6, x = r2 % 6;
      cN = bb * 102400 + y * 40 + x * 2;            // + ci*400 + ky*20 + kx
    }
  }

  v8f acc[4][2];
#pragma unroll
  for (int mi = 0; mi < 4; ++mi)
#pragma unroll
    for (int ni = 0; ni < 2; ++ni)
#pragma unroll
      for (int r = 0; r < 8; ++r) acc[mi][ni][r] = 0.0f;

  auto load_tiles = [&](int kt, int dbuf) {
    // ---- A tile: 256x32 from [256][K] ----
#if HAVE_TDM
    if (wave == 0) {
      unsigned lds_off = (unsigned)(unsigned long long)(&lA[dbuf][0]);
      tdm_load_tile_A(lds_off, A + (size_t)kt * 32, (unsigned)K);
    }
#else
    {
      const unsigned short* src = A + (size_t)tid * K + (size_t)kt * 32;
      uint4 d0 = *(const uint4*)(src);
      uint4 d1 = *(const uint4*)(src + 8);
      uint4 d2 = *(const uint4*)(src + 16);
      uint4 d3 = *(const uint4*)(src + 24);
      uint4* dst = (uint4*)&lA[dbuf][tid * 40];     // byte off = tid*80, 16B aligned
      dst[0] = d0; dst[1] = d1; dst[2] = d2; dst[3] = d3;
    }
#endif
    // ---- B tile: implicit im2col gather, 32k x 64n, stored [n][k] ----
    if (MODE == 0) {
#pragma unroll
      for (int pass = 0; pass < 8; ++pass) {
        const int kk = pass * 4 + khs;              // scalar, 0..31
        const int kg = kt * 32 + kk;                // scalar
        unsigned short val = 0;
        if (kg < 81) {                              // scalar branch
          const int off = (kg / 9) * 28 + (kg % 9); // SALU
          val = Bsrc[cN + off];
        }
        lB[dbuf][nloc * 34 + kk] = val;
      }
    } else {
      // uniform per tile: (ky,kx) fixed, ci = ci0 + kk  (K permuted at build)
      const int kykx = kt >> 3;
      const int ky = kykx / 9, kx = kykx % 9;
      const int base = ky * 20 + kx + ((kt & 7) * 32) * 400;
#pragma unroll
      for (int pass = 0; pass < 8; ++pass) {
        const int kk = pass * 4 + khs;              // scalar
        lB[dbuf][nloc * 34 + kk] = Bsrc[cN + (base + kk * 400)];
      }
    }
  };

  load_tiles(0, 0);
#if HAVE_TDM
  __builtin_amdgcn_s_wait_tensorcnt(0);
#endif
  __syncthreads();

  const int wm = (wave >> 1) * 64;
  const int wn = (wave & 1) * 32;
  const int mrow = lane & 15;
  const int hlf = lane >> 4;

  int buf = 0;
  for (int kt = 0; kt < NSTEPS; ++kt) {
    if (kt + 1 < NSTEPS) load_tiles(kt + 1, buf ^ 1);

    // A fragments: ISA 16-bit A 16x32 layout. kb = (v>>2)*16 + half*8 + (v&3)*2
    v8u au[4];
#pragma unroll
    for (int mi = 0; mi < 4; ++mi) {
      const unsigned short* base = &lA[buf][(wm + mi * 16 + mrow) * 40];
      v8u t;
#pragma unroll
      for (int v = 0; v < 8; ++v) {
        int kb = ((v >> 2) << 4) + (hlf << 3) + ((v & 3) << 1);
        t[v] = *(const unsigned*)(base + kb);
      }
      au[mi] = t;
    }
    // B fragments: same k order as A. kb = (v>>2)*16 + half*8 + (v&3)*2
    v8u bu[2];
#pragma unroll
    for (int ni = 0; ni < 2; ++ni) {
      const unsigned short* base =
          &lB[buf][(wn + ni * 16 + mrow) * 34];
      v8u t;
#pragma unroll
      for (int v = 0; v < 8; ++v) t[v] = *(const unsigned*)(base + (((v >> 2) << 4) + (hlf << 3) + ((v & 3) << 1)));
      bu[ni] = t;
    }
#pragma unroll
    for (int mi = 0; mi < 4; ++mi)
#pragma unroll
      for (int ni = 0; ni < 2; ++ni)
        acc[mi][ni] = __builtin_amdgcn_wmma_f32_16x16x32_bf16(
            false, __builtin_bit_cast(v16bf, au[mi]),
            false, __builtin_bit_cast(v16bf, bu[ni]),
            (short)0, acc[mi][ni], false, false);

#if HAVE_TDM
    __builtin_amdgcn_s_wait_tensorcnt(0);
#endif
    __syncthreads();
    buf ^= 1;
  }

  // Epilogue. D layout: VGPR r -> M = r + 8*half, N = lane&15.
#pragma unroll
  for (int mi = 0; mi < 4; ++mi) {
#pragma unroll
    for (int ni = 0; ni < 2; ++ni) {
      const int gn = n0 + wn + ni * 16 + (lane & 15);
      const int bb = (MODE == 0) ? (gn / 400) : (gn / 36);
      const int yx = (MODE == 0) ? (gn % 400) : (gn % 36);
#pragma unroll
      for (int r = 0; r < 8; ++r) {
        const int gm = wm + mi * 16 + hlf * 8 + r;
        float v = acc[mi][ni][r] + bias[gm];
        if (MODE == 0) {
          v = fmaxf(v, 0.0f);
          ((unsigned short*)Out)[(size_t)(bb * 256 + gm) * 400 + yx] = f2bf(v);
        } else {
          ((float*)Out)[(size_t)bb * 9216 + gm * 36 + yx] = v;
        }
      }
    }
  }
}

// ---------------------------------------------------------------------------
// Squash factors over the 1152-dim axis of u = p.reshape(B,8,1152).
// ---------------------------------------------------------------------------
__global__ __launch_bounds__(256) void squash_norm_kernel(
    const float* __restrict__ p, float* __restrict__ factor) {
  int bu = blockIdx.x;  // b*8 + unit, 0..4095
  const float* src = p + (size_t)bu * 1152;
  __shared__ float red[256];
  float s = 0.0f;
  for (int i = threadIdx.x; i < 1152; i += 256) {
    float v = src[i];
    s += v * v;
  }
  red[threadIdx.x] = s;
  __syncthreads();
  for (int off = 128; off > 0; off >>= 1) {
    if (threadIdx.x < off) red[threadIdx.x] += red[threadIdx.x + off];
    __syncthreads();
  }
  if (threadIdx.x == 0) {
    float sq = red[0];
    factor[bu] = sq / ((1.0f + sq) * sqrtf(sq + 1e-8f));
  }
}

// ---------------------------------------------------------------------------
// u_hat[b,i,j,f] = sum_e u[b,i,e] * W[i,j,e,f],  u[b,i,e]=factor[b,e]*p[b,e*1152+i]
// ---------------------------------------------------------------------------
__global__ __launch_bounds__(160) void uhat_kernel(
    const float* __restrict__ p, const float* __restrict__ factor,
    const float* __restrict__ W, unsigned short* __restrict__ uhat) {
  const int i = blockIdx.x;
  const int b = blockIdx.y;
  __shared__ float ush[8];
  const int t = threadIdx.x;
  if (t < 8) ush[t] = factor[b * 8 + t] * p[(size_t)b * 9216 + t * 1152 + i];
  __syncthreads();
  const int j = t / 16, f = t % 16;
  const float* w = W + ((size_t)(i * 10 + j) * 8) * 16 + f;
  float acc = 0.0f;
#pragma unroll
  for (int e = 0; e < 8; ++e) acc += ush[e] * w[e * 16];
  uhat[((size_t)(b * 1152 + i) * 10 + j) * 16 + f] = f2bf(acc);
}

// ---------------------------------------------------------------------------
// Fused routing iteration (one block per batch element b).
// ---------------------------------------------------------------------------
__global__ __launch_bounds__(256) void routing_kernel(
    const unsigned short* __restrict__ uhat, float* __restrict__ blog,
    float* __restrict__ vout, int iter) {
  const int b = blockIdx.x;
  __shared__ float csh[1152 * 10];
  __shared__ float vsh[160];
  __shared__ float ssh[160];
  __shared__ float fsh[10];
  const int t = threadIdx.x;
  if (iter > 0 && t < 160) vsh[t] = vout[b * 160 + t];
  __syncthreads();

  const unsigned short* ub = uhat + (size_t)b * 1152 * 160;
  float* bb = blog + (size_t)b * 1152 * 10;

  for (int i = t; i < 1152; i += 256) {
    float bl[10];
    if (iter == 0) {
#pragma unroll
      for (int j = 0; j < 10; ++j) {
        bl[j] = 0.0f;
        bb[i * 10 + j] = 0.0f;  // deterministic init for next iteration
      }
    } else {
      const unsigned short* u = ub + (size_t)i * 160;
#pragma unroll
      for (int j = 0; j < 10; ++j) {
        float d = 0.0f;
#pragma unroll
        for (int f = 0; f < 16; ++f) d += bf2f(u[j * 16 + f]) * vsh[j * 16 + f];
        float nb = bb[i * 10 + j] + d;
        bb[i * 10 + j] = nb;
        bl[j] = nb;
      }
    }
    float m = bl[0];
#pragma unroll
    for (int j = 1; j < 10; ++j) m = fmaxf(m, bl[j]);
    float s = 0.0f;
#pragma unroll
    for (int j = 0; j < 10; ++j) {
      bl[j] = expf(bl[j] - m);
      s += bl[j];
    }
    float inv = 1.0f / s;
#pragma unroll
    for (int j = 0; j < 10; ++j) csh[i * 10 + j] = bl[j] * inv;
  }
  __syncthreads();

  if (t < 160) {
    const int j = t / 16;
    float acc = 0.0f;
    for (int i = 0; i < 1152; ++i)
      acc += csh[i * 10 + j] * bf2f(ub[(size_t)i * 160 + t]);
    ssh[t] = acc;
  }
  __syncthreads();
  if (t < 10) {
    float sq = 0.0f;
#pragma unroll
    for (int f = 0; f < 16; ++f) sq += ssh[t * 16 + f] * ssh[t * 16 + f];
    fsh[t] = sq / ((1.0f + sq) * sqrtf(sq + 1e-8f));
  }
  __syncthreads();
  if (t < 160) vout[b * 160 + t] = ssh[t] * fsh[t / 16];
}

// ---------------------------------------------------------------------------
// Host-side launch
// ---------------------------------------------------------------------------
extern "C" void kernel_launch(void* const* d_in, const int* in_sizes, int n_in,
                              void* d_out, int out_size, void* d_ws,
                              size_t ws_size, hipStream_t stream) {
  (void)in_sizes; (void)n_in; (void)out_size; (void)ws_size;
  const float* x  = (const float*)d_in[0];   // [512,1,28,28]
  const float* w1 = (const float*)d_in[1];   // [256,1,9,9]
  const float* b1 = (const float*)d_in[2];   // [256]
  const float* w2 = (const float*)d_in[3];   // [256,256,9,9]
  const float* b2 = (const float*)d_in[4];   // [256]
  const float* Wd = (const float*)d_in[5];   // [1152,10,8,16]
  float* vout = (float*)d_out;               // [512,10,16]

  char* ws = (char*)d_ws;
  size_t off = 0;
  auto take = [&](size_t bytes) -> char* {
    char* q = ws + off;
    off = (off + bytes + 255) & ~(size_t)255;
    return q;
  };
  unsigned short* xbf = (unsigned short*)take((size_t)512 * 784 * 2);
  unsigned short* A1  = (unsigned short*)take((size_t)256 * 96 * 2);
  unsigned short* h   = (unsigned short*)take((size_t)512 * 256 * 400 * 2);
  unsigned short* A2  = (unsigned short*)take((size_t)256 * 20736 * 2);
  float* pbuf         = (float*)take((size_t)512 * 9216 * 4);
  float* factor       = (float*)take((size_t)4096 * 4);
  unsigned short* uh  = (unsigned short*)take((size_t)512 * 1152 * 160 * 2);
  float* blog         = (float*)take((size_t)512 * 1152 * 10 * 4);

  cvt_bf16_kernel<<<(512 * 784 + 255) / 256, 256, 0, stream>>>(x, xbf, 512 * 784);
  build_A1_kernel<<<(256 * 96 + 255) / 256, 256, 0, stream>>>(w1, A1);
  build_A2_kernel<<<(256 * 20736 + 255) / 256, 256, 0, stream>>>(w2, A2);

  // conv1+ReLU: M=256, N=204800, K=96 -> h (bf16)
  gemm_wmma_kernel<0><<<dim3(3200), 256, 0, stream>>>(A1, xbf, b1, (void*)h);
  // conv2: M=256, N=18432, K=20736 -> p (f32, bias added)
  gemm_wmma_kernel<1><<<dim3(288), 256, 0, stream>>>(A2, h, b2, (void*)pbuf);

  squash_norm_kernel<<<4096, 256, 0, stream>>>(pbuf, factor);
  uhat_kernel<<<dim3(1152, 512), 160, 0, stream>>>(pbuf, factor, Wd, uh);

  for (int it = 0; it < 3; ++it)
    routing_kernel<<<512, 256, 0, stream>>>(uh, blog, vout, it);
}
